// HardNegativeMiner_39075612459527
// MI455X (gfx1250) — matrix operand order repaired and checked by reading, not compiled, not run on hardware
//
#include <hip/hip_runtime.h>
#include <hip/hip_bf16.h>
#include <stdint.h>

// ---------------------------------------------------------------------------
// Types for CDNA5 WMMA (wave32): bf16 16x16x32, f32 accumulate.
// ---------------------------------------------------------------------------
typedef __bf16 bf16_t;
typedef bf16_t v16bf __attribute__((ext_vector_type(16)));
typedef float  v8f   __attribute__((ext_vector_type(8)));

#define KNEG 64   // num_negatives (reference setup)
#define NT   4    // N-tiles (16 cols each) per A-fragment reuse (64-col group)
#define KC   64   // K-chunk (bf16 elems) staged per async B buffer

// LDS layout (dynamic shared; offsets are raw LDS byte offsets, no statics)
#define LDS_AF   0u          // 16 x 1024 bf16 = 32768 B (feature A panel)
#define LDS_AL   32768u      // 16 x  128 bf16 =  4096 B (label A panel)
#define LDS_B    36864u      // 2 x 8192 B (B chunk double buffer: 64 rows x 64 bf16)
#define LDS_TOPS 53248u      // 16*64 f32
#define LDS_TOPI 57344u      // 16*64 i32
#define LDS_TILE 61440u      // 16*16 f32
#define LDS_TOTAL 62464u

__device__ __forceinline__ unsigned short f2bf(float f) {
    uint32_t u = __float_as_uint(f);
    u += 0x7fffu + ((u >> 16) & 1u);      // round-to-nearest-even
    return (unsigned short)(u >> 16);
}
__device__ __forceinline__ float bf2f(unsigned short b) {
    return __uint_as_float(((uint32_t)b) << 16);
}

// ---------------------------------------------------------------------------
// CDNA5 async global->LDS copy (ASYNCcnt-tracked), GV addressing mode.
// ---------------------------------------------------------------------------
__device__ __forceinline__ void async_g2l_b128(unsigned lds_off, const void* gptr) {
    asm volatile("global_load_async_to_lds_b128 %0, %1, off"
                 :: "v"(lds_off), "v"((unsigned long long)(uintptr_t)gptr)
                 : "memory");
}
__device__ __forceinline__ void wait_async_16() {
    asm volatile("s_wait_asynccnt 0x10" ::: "memory");
}
__device__ __forceinline__ void wait_async_0() {
    asm volatile("s_wait_asynccnt 0x0" ::: "memory");
}

// ---------------------------------------------------------------------------
// Row L2-normalize (one wave per row), emit bf16 rows + (optional) squared
// norm of the bf16-rounded row (keeps dist diagonal ~0 under bf16 WMMA).
// ---------------------------------------------------------------------------
__global__ __launch_bounds__(256)
void nrm_rows_kernel(const float* __restrict__ in, unsigned short* __restrict__ outb,
                     float* __restrict__ outn, int cols) {
    const int lane = threadIdx.x & 31;
    const int wid  = threadIdx.x >> 5;
    const int row  = blockIdx.x * (blockDim.x >> 5) + wid;

    const float* rp = in + (size_t)row * cols;
    float ss = 0.f;
    for (int c = lane; c < cols; c += 32) { float v = rp[c]; ss += v * v; }
    #pragma unroll
    for (int o = 16; o > 0; o >>= 1) ss += __shfl_xor(ss, o, 32);

    const float inv = 1.f / fmaxf(sqrtf(ss), 1e-12f);

    unsigned short* op = outb + (size_t)row * cols;
    float s2 = 0.f;
    for (int c = lane; c < cols; c += 32) {
        unsigned short b = f2bf(rp[c] * inv);
        op[c] = b;
        float vb = bf2f(b);
        s2 += vb * vb;
    }
    if (outn) {
        #pragma unroll
        for (int o = 16; o > 0; o >>= 1) s2 += __shfl_xor(s2, o, 32);
        if (lane == 0) outn[row] = s2;
    }
}

// ---------------------------------------------------------------------------
// WMMA fragment loaders (contiguous k order shared by A and B, not AMD's table order; wave32).
// A (16x32 bf16): lane m=lane&15, half=lane>>4; elems 0..15 = K[k0+16h..+15].
// B (32x16 bf16): lane n=lane&15, half=lane>>4; elems 0..15 = K[k0+16h..+15].
// ---------------------------------------------------------------------------
union Frag { v16bf v; uint4 u[2]; };

__device__ __forceinline__ v16bf load_a16(const unsigned short* __restrict__ base,
                                          int row0, int k0, int ld, int lane) {
    const int m = lane & 15, h = lane >> 4;
    const unsigned short* p = base + (size_t)(row0 + m) * ld + k0 + 16 * h;
    Frag f;
    f.u[0] = *(const uint4*)(p);
    f.u[1] = *(const uint4*)(p + 8);
    return f.v;
}
__device__ __forceinline__ v16bf load_b16(const unsigned short* __restrict__ base,
                                          int col0, int k0, int ld, int lane) {
    const int n = lane & 15, h = lane >> 4;
    const unsigned short* p = base + (size_t)(col0 + n) * ld + k0 + 16 * h;
    Frag f;
    f.u[0] = *(const uint4*)(p);
    f.u[1] = *(const uint4*)(p + 8);
    return f.v;
}
// B fragment out of an LDS chunk laid out [64 rows][KC bf16], row-major.
__device__ __forceinline__ v16bf load_b_lds(const unsigned short* chunk,
                                            int t, int kk, int lane) {
    const int n = lane & 15, h = lane >> 4;
    const unsigned short* p = chunk + (size_t)(16 * t + n) * KC + kk + 16 * h;
    Frag f;
    f.u[0] = *(const uint4*)(p);
    f.u[1] = *(const uint4*)(p + 8);
    return f.v;
}

// ---------------------------------------------------------------------------
// Fused Gram + score + streaming top-64 miner. One wave per 16-row M-panel.
// A panel LDS-resident; B panel async double-buffered through LDS.
// C/D layout: VGPR e, lane L: M = e + 8*(L>=16), N = L&15.
// ---------------------------------------------------------------------------
__global__ __launch_bounds__(32)
void mine_topk_kernel(const unsigned short* __restrict__ fnorm,
                      const unsigned short* __restrict__ lnorm,
                      const float* __restrict__ nf,
                      int* __restrict__ outIdx, float* __restrict__ outScore,
                      int B, int D, int T) {
    extern __shared__ char smem[];
    unsigned short* sAf   = (unsigned short*)(smem + LDS_AF);
    unsigned short* sAl   = (unsigned short*)(smem + LDS_AL);
    float*          sTopS = (float*)(smem + LDS_TOPS);
    int*            sTopI = (int*)(smem + LDS_TOPI);
    float*          sTile = (float*)(smem + LDS_TILE);

    const int lane = threadIdx.x;         // 32 threads = 1 wave
    const int m0   = blockIdx.x * 16;
    const int half = lane >> 4, nn = lane & 15;

    for (int i = lane; i < 16 * KNEG; i += 32) { sTopS[i] = 0.f; sTopI[i] = 0; }

    // ---- Stage the 16-row A panels (features 32KB, labels 4KB) via async DMA.
    for (int o = 0; o < 16 * 1024 * 2; o += 512) {       // feature rows: 2048B each
        const int oo = o + lane * 16;
        const int r  = oo >> 11, cb = oo & 2047;
        async_g2l_b128(LDS_AF + oo, (const char*)(fnorm + (size_t)(m0 + r) * D) + cb);
    }
    for (int o = 0; o < 16 * 128 * 2; o += 512) {        // label rows: 256B each
        const int oo = o + lane * 16;
        const int r  = oo >> 8, cb = oo & 255;
        async_g2l_b128(LDS_AL + oo, (const char*)(lnorm + (size_t)(m0 + r) * T) + cb);
    }
    wait_async_0();
    __syncthreads();

    // Per-lane row norms for the 8 accumulator rows this lane owns.
    float nfr[8];
    #pragma unroll
    for (int e = 0; e < 8; ++e) nfr[e] = nf[m0 + e + 8 * half];

    float curMin = 0.f; int curMinPos = 0;

    const v8f vzero = {0.f, 0.f, 0.f, 0.f, 0.f, 0.f, 0.f, 0.f};
    const int ngroups = B / (16 * NT);
    const int nchunks = D / KC;                           // 16

    for (int g = 0; g < ngroups; ++g) {
        const int n0 = g * 16 * NT;

        if (g + 1 < ngroups) {   // warm next group's label rows toward the WGP
            __builtin_prefetch(lnorm + (size_t)(n0 + 16 * NT + lane) * T, 0, 1);
            __builtin_prefetch(lnorm + (size_t)(n0 + 16 * NT + 32 + lane) * T, 0, 1);
        }

        v8f accS[NT], accG[NT];
        #pragma unroll
        for (int t = 0; t < NT; ++t) { accS[t] = vzero; accG[t] = vzero; }

        // Label-similarity Gram: K = T (128), A from LDS, B from global/L2.
        for (int k0 = 0; k0 < T; k0 += 32) {
            v16bf aL = load_a16(sAl, 0, k0, T, lane);
            #pragma unroll
            for (int t = 0; t < NT; ++t) {
                v16bf bL = load_b16(lnorm, n0 + 16 * t, k0, T, lane);
                accS[t] = __builtin_amdgcn_wmma_f32_16x16x32_bf16(
                    false, aL, false, bL, (short)0, accS[t], false, false);
            }
        }

        // Feature Gram: K = D (1024), async double-buffered B chunks in LDS.
        // Stage chunk 0 (64 rows x 128B = 16 x b128-per-lane).
        #pragma unroll
        for (int o = 0; o < 64 * KC * 2; o += 512) {
            const int oo = o + lane * 16;
            const int r  = oo >> 7, cb = oo & 127;
            async_g2l_b128(LDS_B + oo, (const char*)(fnorm + (size_t)(n0 + r) * D) + cb);
        }
        for (int c = 0; c < nchunks; ++c) {
            if (c + 1 < nchunks) {
                const unsigned dst = LDS_B + ((unsigned)(c + 1) & 1u) * 8192u;
                const int k1 = (c + 1) * KC;
                #pragma unroll
                for (int o = 0; o < 64 * KC * 2; o += 512) {
                    const int oo = o + lane * 16;
                    const int r  = oo >> 7, cb = oo & 127;
                    async_g2l_b128(dst + oo,
                                   (const char*)(fnorm + (size_t)(n0 + r) * D + k1) + cb);
                }
                wait_async_16();     // chunk c landed; chunk c+1 still in flight
            } else {
                wait_async_0();
            }
            const unsigned short* chunk =
                (const unsigned short*)(smem + LDS_B + ((unsigned)c & 1u) * 8192u);
            #pragma unroll
            for (int kk = 0; kk < KC; kk += 32) {
                v16bf aF = load_a16(sAf, 0, c * KC + kk, D, lane);
                #pragma unroll
                for (int t = 0; t < NT; ++t) {
                    v16bf bF = load_b_lds(chunk, t, kk, lane);
                    accG[t] = __builtin_amdgcn_wmma_f32_16x16x32_bf16(
                        false, aF, false, bF, (short)0, accG[t], false, false);
                }
            }
        }

        // Scores + streaming top-64 update, one 16x16 tile at a time.
        for (int t = 0; t < NT; ++t) {
            const int c0  = n0 + 16 * t;
            const int col = c0 + nn;
            const float nfc = nf[col];
            #pragma unroll
            for (int e = 0; e < 8; ++e) {
                const int rowM = e + 8 * half;
                const float gg = accG[t][e];
                const float ss = accS[t][e];
                const float sq = fmaxf(nfr[e] + nfc - 2.f * gg, 0.f);
                const float d  = sqrtf(sq);
                const int row  = m0 + rowM;
                const bool msk = (ss < 0.3f) && (d < 1.45f) && (row != col);
                const float sc = msk ? (1.f - ss) / (d + 1e-8f) : 0.f;
                sTile[rowM * 16 + nn] = sc;
            }
            __syncthreads();
            if (lane < 16) {                 // lane r owns row r's top-64 list
                const int r = lane;
                for (int j = 0; j < 16; ++j) {
                    const float sc = sTile[r * 16 + j];
                    if (sc > curMin) {
                        sTopS[r * KNEG + curMinPos] = sc;
                        sTopI[r * KNEG + curMinPos] = c0 + j;
                        float mn = sTopS[r * KNEG]; int mp = 0;
                        for (int q = 1; q < KNEG; ++q) {
                            const float v = sTopS[r * KNEG + q];
                            if (v < mn) { mn = v; mp = q; }
                        }
                        curMin = mn; curMinPos = mp;
                    }
                }
            }
            __syncthreads();
        }
    }

    // Sort each row's 64 entries descending (owner lane, insertion sort).
    if (lane < 16) {
        const int r = lane;
        for (int i = 1; i < KNEG; ++i) {
            const float sv = sTopS[r * KNEG + i];
            const int   iv = sTopI[r * KNEG + i];
            int j = i - 1;
            while (j >= 0 && sTopS[r * KNEG + j] < sv) {
                sTopS[r * KNEG + j + 1] = sTopS[r * KNEG + j];
                sTopI[r * KNEG + j + 1] = sTopI[r * KNEG + j];
                --j;
            }
            sTopS[r * KNEG + j + 1] = sv;
            sTopI[r * KNEG + j + 1] = iv;
        }
    }
    __syncthreads();

    for (int i = lane; i < 16 * KNEG; i += 32) {
        const int r = i / KNEG, t = i % KNEG;
        outIdx[(size_t)(m0 + r) * KNEG + t]   = sTopI[i];
        outScore[(size_t)(m0 + r) * KNEG + t] = sTopS[i];
    }
}

// ---------------------------------------------------------------------------
// Launch: normalize -> bf16 workspace, then fused WMMA miner.
// d_out = [top_idx int32 B*64][top_scores f32 B*64]  (both 4B elements).
// ---------------------------------------------------------------------------
extern "C" void kernel_launch(void* const* d_in, const int* in_sizes, int n_in,
                              void* d_out, int out_size, void* d_ws, size_t ws_size,
                              hipStream_t stream) {
    const float* features = (const float*)d_in[0];
    const float* labels   = (const float*)d_in[1];
    (void)n_in; (void)out_size; (void)ws_size;

    const int D = 1024;
    const int T = 128;
    const int B = in_sizes[0] / D;        // 8192

    unsigned short* fnorm = (unsigned short*)d_ws;
    unsigned short* lnorm = (unsigned short*)((char*)d_ws + (size_t)B * D * sizeof(unsigned short));
    float* nf = (float*)((char*)d_ws + (size_t)B * D * sizeof(unsigned short)
                                     + (size_t)B * T * sizeof(unsigned short));

    int*   outIdx   = (int*)d_out;
    float* outScore = (float*)d_out + (size_t)B * KNEG;

    nrm_rows_kernel<<<B / 8, 256, 0, stream>>>(features, fnorm, nf, D);
    nrm_rows_kernel<<<B / 8, 256, 0, stream>>>(labels, lnorm, nullptr, T);
    mine_topk_kernel<<<B / 16, 32, LDS_TOTAL, stream>>>(fnorm, lnorm, nf,
                                                        outIdx, outScore, B, D, T);
}
